// SparseMoE_41540923687611
// MI455X (gfx1250) — matrix operand order repaired and checked by reading, not compiled, not run on hardware
//
#include <hip/hip_runtime.h>

#define D_EMB   1024
#define N_EXP   8
#define N_ROUT  5
#define SEQLEN  2048
#define H_DIM   4096
#define NTOK    (N_ROUT * SEQLEN)          /* 10240 tokens */
#define CAP     NTOK                        /* worst-case tokens per expert */
#define TILES_PER_E ((CAP + 15) / 16)       /* 640 */
#define OUT_F   ((size_t)NTOK * D_EMB)      /* 10,485,760 floats */
#define HC      128                         /* H-chunk per fused step */

#define KS1 (D_EMB / 32)                    /* 32  k-steps, layer 1 */
#define NT1 (H_DIM / 16)                    /* 256 n-tiles, layer 1 */
#define KS2 (H_DIM / 32)                    /* 128 k-steps, layer 2 */
#define NT2 (D_EMB / 16)                    /* 64  n-tiles, layer 2 */
#define TILE_U16 512                        /* 32 lanes x 16 bf16 per B-tile */

typedef __attribute__((ext_vector_type(16))) __bf16 v16bf;
typedef __attribute__((ext_vector_type(8)))  float  v8f;
typedef __attribute__((ext_vector_type(16))) unsigned short v16u;
typedef __attribute__((ext_vector_type(4)))  int    v4i;

#if defined(__has_builtin)
#  if __has_builtin(__builtin_amdgcn_global_load_async_to_lds_b128)
#    define HAVE_ASYNC_LDS 1
#  endif
#endif

__device__ __forceinline__ unsigned short f2bf(float f) {
  return __builtin_bit_cast(unsigned short, (__bf16)f);   // native cvt if available
}
__device__ __forceinline__ v16bf as_bf16x16(v16u v) {
  return __builtin_bit_cast(v16bf, v);
}

// 16B global -> LDS copy; async DMA (ASYNCcnt) on gfx1250 toolchains that expose it
__device__ __forceinline__ void copy_b128_to_lds(const float* g, float* lds) {
#ifdef HAVE_ASYNC_LDS
  typedef __attribute__((address_space(1))) v4i g_v4i;    // global b128 payload
  typedef __attribute__((address_space(3))) v4i l_v4i;    // LDS b128 payload
  __builtin_amdgcn_global_load_async_to_lds_b128((g_v4i*)g, (l_v4i*)lds, 0, 0);
#else
  *(float4*)lds = *(const float4*)g;
#endif
}
__device__ __forceinline__ void async_drain() {
#ifdef HAVE_ASYNC_LDS
#  if __has_builtin(__builtin_amdgcn_s_wait_asynccnt)
  __builtin_amdgcn_s_wait_asynccnt(0);
#  else
  asm volatile("s_wait_asynccnt 0" ::: "memory");
#  endif
#endif
}

__device__ __forceinline__ unsigned pcg_hash(unsigned v) {
  v = v * 747796405u + 2891336453u;
  unsigned w = ((v >> ((v >> 28u) + 4u)) ^ v) * 277803737u;
  return (w >> 22u) ^ w;
}
__device__ __forceinline__ float gauss_eps(unsigned idx) {
  unsigned a = pcg_hash(idx * 2u + 1u);
  unsigned b = pcg_hash(idx * 2u + 2u);
  float u1 = ((float)a + 1.0f) * (1.0f / 4294967296.0f);
  float u2 = (float)b * (1.0f / 4294967296.0f);
  return sqrtf(-2.0f * __logf(u1)) * __cosf(6.28318530718f * u2);
}

// ---------------------------------------------------------------- zero
__global__ void zero_out(float* __restrict__ out, int* __restrict__ counts) {
  const size_t i = (size_t)blockIdx.x * blockDim.x + threadIdx.x;
  const size_t stride = (size_t)gridDim.x * blockDim.x;
  for (size_t j = i; j < OUT_F; j += stride) out[j] = 0.f;
  if (blockIdx.x == 0 && threadIdx.x < 16) counts[threadIdx.x] = 0;
}

// ---------------------------------------------------------------- weight swizzle
// One wave per 32x16 B-tile: fp32 -> bf16, stored lane-major so the GEMM
// reads one aligned 32B fragment per lane (2x global_load_b128, no VALU).
__global__ __launch_bounds__(256) void conv_w(
    const float* __restrict__ W1, const float* __restrict__ W2,
    unsigned short* __restrict__ W1s, unsigned short* __restrict__ W2s)
{
  const int lane = threadIdx.x & 31;
  const int ln16 = lane & 15;
  const int hh   = lane >> 4;
  const int w    = blockIdx.x * 8 + (threadIdx.x >> 5);   // global wave id
  const int T1   = N_EXP * KS1 * NT1;                     // 65536 W1 tiles
  const int T2   = N_EXP * KS2 * NT2;                     // 65536 W2 tiles

  if (w < T1) {
    const int e     = w / (KS1 * NT1);
    const int rem   = w % (KS1 * NT1);
    const int k0    = (rem / NT1) * 32;
    const int n0    = (rem % NT1) * 16;
    const float* src = W1 + (size_t)e * D_EMB * H_DIM;
    v16u frag;
    #pragma unroll
    for (int j = 0; j < 16; ++j)
      frag[j] = f2bf(src[(size_t)(k0 + hh * 8 + (j & 7) + ((j >> 3) << 4)) * H_DIM + n0 + ln16]);
    *(v16u*)&W1s[((size_t)w * 32 + lane) * 16] = frag;
  } else if (w < T1 + T2) {
    const int w2    = w - T1;
    const int e     = w2 / (KS2 * NT2);
    const int rem   = w2 % (KS2 * NT2);
    const int k0    = (rem / NT2) * 32;
    const int n0    = (rem % NT2) * 16;
    const float* src = W2 + (size_t)e * H_DIM * D_EMB;
    v16u frag;
    #pragma unroll
    for (int j = 0; j < 16; ++j)
      frag[j] = f2bf(src[(size_t)(k0 + hh * 8 + (j & 7) + ((j >> 3) << 4)) * D_EMB + n0 + ln16]);
    *(v16u*)&W2s[((size_t)w2 * 32 + lane) * 16] = frag;
  }
}

// ---------------------------------------------------------------- router
__global__ __launch_bounds__(256) void moe_router(
    const float* __restrict__ x, const int* __restrict__ moe,
    const float* __restrict__ Wr, const float* __restrict__ br,
    const float* __restrict__ Wn, const float* __restrict__ bn,
    int* __restrict__ topi_out, int* __restrict__ counts,
    int* __restrict__ tok_list, float* __restrict__ gate_list)
{
  const int wave = threadIdx.x >> 5;
  const int lane = threadIdx.x & 31;
  const int t = blockIdx.x * 8 + wave;
  if (t >= NTOK) return;
  const int b = t / SEQLEN;
  const int r = moe[b];
  const float* xr = x + (size_t)t * D_EMB;
  const float* wr = Wr + (size_t)r * D_EMB * N_EXP;
  const float* wn = Wn + (size_t)r * D_EMB * N_EXP;

  float aL[N_EXP], aN[N_EXP];
  #pragma unroll
  for (int e = 0; e < N_EXP; ++e) { aL[e] = 0.f; aN[e] = 0.f; }
  for (int i = lane; i < D_EMB; i += 32) {
    const float xv = xr[i];
    const float* p = wr + i * N_EXP;
    const float* q = wn + i * N_EXP;
    #pragma unroll
    for (int e = 0; e < N_EXP; ++e) {
      aL[e] = fmaf(xv, p[e], aL[e]);
      aN[e] = fmaf(xv, q[e], aN[e]);
    }
  }
  #pragma unroll
  for (int off = 16; off > 0; off >>= 1) {
    #pragma unroll
    for (int e = 0; e < N_EXP; ++e) {
      aL[e] += __shfl_xor(aL[e], off, 32);
      aN[e] += __shfl_xor(aN[e], off, 32);
    }
  }
  if (lane == 0) {
    float noisy[N_EXP];
    #pragma unroll
    for (int e = 0; e < N_EXP; ++e) {
      float lg = aL[e] + br[r * N_EXP + e];
      float nl = aN[e] + bn[r * N_EXP + e];
      float sp = (nl > 20.f) ? nl : log1pf(__expf(nl));
      noisy[e] = lg + gauss_eps((unsigned)(t * N_EXP + e)) * sp;
    }
    int i0 = 0;
    #pragma unroll
    for (int e = 1; e < N_EXP; ++e) if (noisy[e] > noisy[i0]) i0 = e;
    int i1 = (i0 == 0) ? 1 : 0;
    #pragma unroll
    for (int e = 0; e < N_EXP; ++e) if (e != i0 && noisy[e] > noisy[i1]) i1 = e;
    const float v0 = noisy[i0], v1 = noisy[i1];
    const float g0 = 1.f / (1.f + __expf(v1 - v0));
    const float g1 = 1.f - g0;
    topi_out[t * 2 + 0] = i0;
    topi_out[t * 2 + 1] = i1;
    int p0 = atomicAdd(&counts[i0], 1);
    tok_list[i0 * CAP + p0] = t;  gate_list[i0 * CAP + p0] = g0;
    int p1 = atomicAdd(&counts[i1], 1);
    tok_list[i1 * CAP + p1] = t;  gate_list[i1 * CAP + p1] = g1;
  }
}

// ---------------------------------------------------------------- expert FFN
// Block = 16 gathered tokens of one expert, 8 waves. X rows staged into LDS
// by async DMA, then converted into A-fragment order; weights pre-swizzled bf16.
__global__ __launch_bounds__(256, 1) void moe_ffn(
    const float* __restrict__ x,
    const unsigned short* __restrict__ W1s, const float* __restrict__ b1,
    const unsigned short* __restrict__ W2s, const float* __restrict__ b2,
    const int* __restrict__ counts, const int* __restrict__ tok_list,
    const float* __restrict__ gate_list, float* __restrict__ out)
{
  __shared__ __align__(32) unsigned short XsA[KS1 * TILE_U16];       // 32 KB
  __shared__ __align__(32) unsigned short HsA[(HC / 32) * TILE_U16]; // 4 KB
  __shared__ __align__(16) float Xstage[4][D_EMB];                   // 16 KB
  __shared__ int   toks[16];
  __shared__ float gts[16];

  const int e    = blockIdx.x / TILES_PER_E;     // expert-major => L2 reuse of W
  const int tile = blockIdx.x % TILES_PER_E;
  const int cnt  = counts[e];
  const int row0 = tile * 16;
  if (row0 >= cnt) return;

  const int tid  = threadIdx.x;
  const int wave = tid >> 5;
  const int lane = tid & 31;
  const int ln16 = lane & 15;
  const int hi   = lane >> 4;

  if (tid < 16) {
    const int rr = row0 + tid;
    if (rr < cnt) { toks[tid] = tok_list[e * CAP + rr]; gts[tid] = gate_list[e * CAP + rr]; }
    else          { toks[tid] = tok_list[e * CAP];      gts[tid] = 0.f; }   // pad: gate 0
  }
  __syncthreads();

  // ---- gather: 4 rows per phase, async global->LDS, then cvt+swizzle to
  // A-fragment order: kstep=k/32, rem=k%32 -> h=(rem>>3)&1, j=(rem&7)+(rem&16?8:0)
  for (int grp = 0; grp < 4; ++grp) {
    #pragma unroll
    for (int it = 0; it < 4; ++it) {
      const int u    = it * 256 + tid;       // 1024 16B-units = 4 rows x 4KB
      const int rrow = u >> 8;
      const int cu   = u & 255;
      copy_b128_to_lds(x + (size_t)toks[grp * 4 + rrow] * D_EMB + cu * 4,
                       &Xstage[rrow][cu * 4]);
    }
    async_drain();
    __syncthreads();
    for (int li = tid; li < 4 * D_EMB; li += 256) {
      const int mr = li >> 10, k = li & (D_EMB - 1);
      const int m  = grp * 4 + mr;
      const float v = Xstage[mr][k];
      const int rem = k & 31;
      const int h = (rem >> 3) & 1;
      const int j = (rem & 7) + ((rem & 16) ? 8 : 0);
      XsA[(((k >> 5) << 5) + (m + 16 * h)) * 16 + j] = f2bf(v);
    }
    __syncthreads();
  }

  const float* bb1 = b1 + (size_t)e * H_DIM;
  const float* bb2 = b2 + (size_t)e * D_EMB;

  v8f acc2[8] = {};   // this wave's 16 x 128 slice of the 16 x 1024 output

  for (int h0 = 0; h0 < H_DIM; h0 += HC) {
    // ---- layer 1: wave owns cols [h0+wave*16, +16), K = 1024 (32 WMMA steps)
    v8f acc1 = {};
    const int nt1 = (h0 >> 4) + wave;
    const unsigned short* bp1 =
        W1s + ((size_t)(e * KS1 * NT1 + nt1) * 32 + lane) * 16;
    const unsigned short* ap1 = XsA + lane * 16;
    #pragma unroll 4
    for (int ks = 0; ks < KS1; ++ks) {
      const v16u au = *(const v16u*)ap1;               // 2x ds_load_b128
      const v16u bu = *(const v16u*)bp1;               // 2x global_load_b128
      acc1 = __builtin_amdgcn_wmma_f32_16x16x32_bf16(
          false, as_bf16x16(au), false, as_bf16x16(bu), (short)0, acc1, false, false);
      ap1 += TILE_U16;
      bp1 += (size_t)NT1 * TILE_U16;
      __builtin_prefetch(bp1, 0, 1);
    }
    // bias + relu + bf16, swizzled into layer-2 A-fragment order
    const float bias1 = bb1[h0 + wave * 16 + ln16];
    const int kk  = wave * 16 + ln16;                  // h-index within chunk
    const int rem = kk & 31;
    const int h2  = (rem >> 3) & 1;
    const int j2  = (rem & 7) + ((rem & 16) ? 8 : 0);
    #pragma unroll
    for (int v = 0; v < 8; ++v) {
      const float hval = fmaxf(acc1[v] + bias1, 0.f);
      const int m = hi * 8 + v;
      HsA[(((kk >> 5) << 5) + (m + 16 * h2)) * 16 + j2] = f2bf(hval);
    }
    __syncthreads();

    // ---- layer 2: acc2 += h(16x128) @ W2[h0:h0+128, wave's 128 cols]
    const unsigned short* ap2 = HsA + lane * 16;
    const unsigned short* bp2 =
        W2s + ((size_t)(e * KS2 * NT2 + (h0 >> 5) * NT2 + wave * 8) * 32 + lane) * 16;
    #pragma unroll
    for (int k2 = 0; k2 < HC / 32; ++k2) {
      const v16u au = *(const v16u*)ap2;
      const v16bf afrag = as_bf16x16(au);
      #pragma unroll
      for (int nt = 0; nt < 8; ++nt) {
        const v16u bu = *(const v16u*)(bp2 + (size_t)nt * TILE_U16);
        acc2[nt] = __builtin_amdgcn_wmma_f32_16x16x32_bf16(
            false, afrag, false, as_bf16x16(bu), (short)0, acc2[nt], false, false);
      }
      ap2 += TILE_U16;
      bp2 += (size_t)NT2 * TILE_U16;
    }
    __syncthreads();
  }

  // epilogue: + b2, scale by gate, accumulate (token is in exactly 2 expert lists)
  #pragma unroll
  for (int nt = 0; nt < 8; ++nt) {
    const int col2 = (wave * 8 + nt) * 16 + ln16;
    const float bias2 = bb2[col2];
    #pragma unroll
    for (int v = 0; v < 8; ++v) {
      const int m = hi * 8 + v;                  // C-layout row for this lane half
      const float contrib = (acc2[nt][v] + bias2) * gts[m];
      atomicAdd(&out[(size_t)toks[m] * D_EMB + col2], contrib);
    }
  }
}

// ---------------------------------------------------------------- launch
extern "C" void kernel_launch(void* const* d_in, const int* in_sizes, int n_in,
                              void* d_out, int out_size, void* d_ws, size_t ws_size,
                              hipStream_t stream) {
  (void)in_sizes; (void)n_in; (void)out_size; (void)ws_size;
  const float* x   = (const float*)d_in[0];
  const int*   moe = (const int*)  d_in[1];
  const float* Wr  = (const float*)d_in[2];
  const float* br  = (const float*)d_in[3];
  const float* Wn  = (const float*)d_in[4];
  const float* bn  = (const float*)d_in[5];
  const float* W1  = (const float*)d_in[6];
  const float* b1  = (const float*)d_in[7];
  const float* W2  = (const float*)d_in[8];
  const float* b2  = (const float*)d_in[9];

  float* out  = (float*)d_out;
  int*   topi = (int*)((float*)d_out + OUT_F);   // tuple: (final_output, topi)

  // workspace layout
  int*   counts    = (int*)d_ws;                         // 16 ints
  int*   tok_list  = counts + 16;                        // [8][CAP]
  float* gate_list = (float*)(tok_list + N_EXP * CAP);   // [8][CAP]
  size_t head = 64 + (size_t)N_EXP * CAP * 8;
  head = (head + 511) & ~(size_t)511;
  unsigned short* W1s = (unsigned short*)((char*)d_ws + head);          // 67 MB
  unsigned short* W2s = W1s + (size_t)N_EXP * KS1 * NT1 * TILE_U16;     // 67 MB

  zero_out<<<2048, 256, 0, stream>>>(out, counts);
  conv_w<<<(N_EXP * KS1 * NT1 + N_EXP * KS2 * NT2) / 8, 256, 0, stream>>>(
      W1, W2, W1s, W2s);
  moe_router<<<(NTOK + 7) / 8, 256, 0, stream>>>(
      x, moe, Wr, br, Wn, bn, topi, counts, tok_list, gate_list);
  moe_ffn<<<N_EXP * TILES_PER_E, 256, 0, stream>>>(
      x, W1s, b1, W2s, b2, counts, tok_list, gate_list, out);
}
